// WindowAttention_62886911148176
// MI455X (gfx1250) — matrix operand order repaired and checked by reading, not compiled, not run on hardware
//
#include <hip/hip_runtime.h>
#include <hip/hip_bf16.h>

typedef __attribute__((ext_vector_type(16))) __bf16 v16bf;
typedef __attribute__((ext_vector_type(8)))  float  v8f;

#define NTOK   64     // 8x8 tokens per window
#define CDIM   256
#define TC     768    // 3*CDIM
#define HEADS  8
#define HD     32
#define SCALE  0.17677669529663687f  // 32^-0.5

// ---------------- fragment helpers (ISA 7.12.2 layouts, wave32) -------------
// A (16x32 bf16): lane m=lane&15, element e -> K = (e>>3)*16 + hf*8 + (e&7)
static __device__ __forceinline__ v16bf load_a_frag(const __bf16* p, int ld, int kbase, int lane) {
    const int m = lane & 15, hf = lane >> 4;
    const __bf16* a0 = p + m * ld + kbase + hf * 8;
    const __bf16* a1 = p + m * ld + kbase + 16 + hf * 8;
    union { v16bf v; uint4 u[2]; } r;
    r.u[0] = *(const uint4*)a0;
    r.u[1] = *(const uint4*)a1;
    return r.v;
}
// B (32x16 bf16) sourced from row-major-in-K storage (i.e. B^T rows):
// lane col n=lane&15, element e -> K = (e>>3)*16 + hf*8 + (e&7)
static __device__ __forceinline__ v16bf load_b_frag(const __bf16* p, int ld, int kbase, int lane) {
    const int n = lane & 15, hf = lane >> 4;
    const __bf16* b0 = p + n * ld + kbase + hf * 8;
    union { v16bf v; uint4 u[2]; } r;
    r.u[0] = *(const uint4*)b0;
    r.u[1] = *(const uint4*)(b0 + 16);
    return r.v;
}
static __device__ __forceinline__ v8f wmma_bf16(v16bf a, v16bf b, v8f c) {
    return __builtin_amdgcn_wmma_f32_16x16x32_bf16(false, a, false, b, (short)0, c, false, false);
}
static __device__ __forceinline__ v8f vzero8() {
    v8f z;
    #pragma unroll
    for (int i = 0; i < 8; ++i) z[i] = 0.0f;
    return z;
}

// ---------------- prep: weights->bf16, build [h][n][m] rel-pos bias ---------
__global__ void prep_kernel(const float* __restrict__ qkvw,   // [768,256]
                            const float* __restrict__ projw,  // [256,256]
                            const float* __restrict__ rpb,    // [225,8]
                            __bf16* __restrict__ wq, __bf16* __restrict__ wp,
                            float* __restrict__ biasb) {      // [8,64,64]
    int i = blockIdx.x * 256 + threadIdx.x;
    if (i < TC * CDIM)   wq[i] = (__bf16)qkvw[i];
    if (i < CDIM * CDIM) wp[i] = (__bf16)projw[i];
    if (i < HEADS * NTOK * NTOK) {
        int h = i >> 12, n = (i >> 6) & 63, m = i & 63;
        int dy = (n >> 3) - (m >> 3) + 7;
        int dx = (n & 7) - (m & 7) + 7;
        biasb[i] = rpb[(dy * 15 + dx) * HEADS + h];
    }
}

// ---------------- fused window-attention kernel -----------------------------
__global__ __launch_bounds__(256, 1)
void winattn_kernel(const float* __restrict__ x,      // [B,8,8,256]
                    const float* __restrict__ qkvb,   // [768]
                    const float* __restrict__ dww,    // [3,3,1,768]
                    const float* __restrict__ dwb,    // [768]
                    const float* __restrict__ temp,   // [8]
                    const float* __restrict__ projb,  // [256]
                    const __bf16* __restrict__ wq,    // [768,256] bf16
                    const __bf16* __restrict__ wp,    // [256,256] bf16
                    const float* __restrict__ biasb,  // [8,64,64]
                    float* __restrict__ outp) {       // [B,64,256]
    extern __shared__ char smem[];
    __bf16* s_x   = (__bf16*)smem;                       // 32 KB: x bf16 / attn-out bf16
    char*   regQ  = smem + 32768;                        // 96 KB: qkv pre-conv / S heads 0-5
    char*   regK  = smem + 32768 + 98304;                // 96 KB: q|k (64K) + vT (32K); q|k -> P
    char*   regS2 = smem + 32768 + 98304 + 98304;        // 32 KB: S heads 6-7
    __bf16* s_qkv = (__bf16*)regQ;                       // [64][768]
    __bf16* s_qk  = (__bf16*)regK;                       // [64][512]  (q | k, normalized)
    __bf16* s_vT  = (__bf16*)(regK + 65536);             // [256][64]  (v transposed)

    const int tid  = threadIdx.x;
    const int wave = tid >> 5;
    const int lane = tid & 31;
    const int hf   = lane >> 4;
    const int cl   = lane & 15;
    const int blk  = blockIdx.x;

    // ---- load x tile, convert to bf16 (64 floats/thread, contiguous) ----
    {
        const float* px = x + (size_t)blk * (NTOK * CDIM) + tid * 64;
        #pragma unroll
        for (int j = 0; j < 64; j += 4) {
            float4 t = *(const float4*)(px + j);
            __bf16* d = s_x + tid * 64 + j;
            d[0] = (__bf16)t.x; d[1] = (__bf16)t.y;
            d[2] = (__bf16)t.z; d[3] = (__bf16)t.w;
        }
    }
    __syncthreads();

    // ---- GEMM1: qkv[64,768] = x[64,256] @ wq^T, + qkv_b, -> s_qkv bf16 ----
    {
        const int nt0 = wave * 6;
        for (int nt = nt0; nt < nt0 + 6; ++nt) {
            if (nt + 1 < nt0 + 6) __builtin_prefetch(wq + (nt + 1) * 16 * CDIM, 0, 3);
            float bcol = qkvb[nt * 16 + cl];
            v8f acc[4];
            #pragma unroll
            for (int mt = 0; mt < 4; ++mt) acc[mt] = vzero8();
            #pragma unroll
            for (int kk = 0; kk < 8; ++kk) {
                v16bf b = load_b_frag(wq + nt * 16 * CDIM, CDIM, kk * 32, lane);
                #pragma unroll
                for (int mt = 0; mt < 4; ++mt) {
                    v16bf a = load_a_frag(s_x + mt * 16 * CDIM, CDIM, kk * 32, lane);
                    acc[mt] = wmma_bf16(a, b, acc[mt]);
                }
            }
            #pragma unroll
            for (int mt = 0; mt < 4; ++mt) {
                #pragma unroll
                for (int r = 0; r < 8; ++r) {
                    int row = mt * 16 + r + hf * 8;
                    s_qkv[row * TC + nt * 16 + cl] = (__bf16)(acc[mt][r] + bcol);
                }
            }
        }
    }
    __syncthreads();

    // ---- depthwise 3x3 conv (+dw_b), then q/k L2-norm; q,k->s_qk, v->s_vT ----
    {
        for (int it = 0; it < 6; ++it) {
            int gid   = it * 256 + tid;           // 1536 groups = 64 tok * 3 mat * 8 heads
            int token = gid / 24;
            int rem   = gid - token * 24;
            int m3    = rem >> 3, h = rem & 7;
            int cb    = m3 * 256 + h * 32;
            int ty = token >> 3, tx = token & 7;
            float acc[HD];
            #pragma unroll
            for (int d = 0; d < HD; ++d) acc[d] = dwb[cb + d];
            #pragma unroll
            for (int dy = -1; dy <= 1; ++dy) {
                int ny = ty + dy;
                if ((unsigned)ny >= 8u) continue;
                #pragma unroll
                for (int dx = -1; dx <= 1; ++dx) {
                    int nx = tx + dx;
                    if ((unsigned)nx >= 8u) continue;
                    const __bf16* src = s_qkv + (ny * 8 + nx) * TC + cb;
                    const float*  w   = dww + ((dy + 1) * 3 + (dx + 1)) * TC + cb;
                    #pragma unroll
                    for (int d = 0; d < HD; ++d) acc[d] += (float)src[d] * w[d];
                }
            }
            if (m3 == 2) {
                #pragma unroll
                for (int d = 0; d < HD; ++d)
                    s_vT[(h * HD + d) * NTOK + token] = (__bf16)acc[d];
            } else {
                float ss = 0.0f;
                #pragma unroll
                for (int d = 0; d < HD; ++d) ss += acc[d] * acc[d];
                float f = 1.0f / fmaxf(sqrtf(ss), 1e-12f);
                if (m3 == 0) f *= SCALE * temp[h];    // fold scale & temperature into q
                #pragma unroll
                for (int d = 0; d < HD; ++d)
                    s_qk[token * 512 + m3 * 256 + h * 32 + d] = (__bf16)(acc[d] * f);
            }
        }
    }
    __syncthreads();

    // ---- per-head (head = wave): S = q k^T + bias, one WMMA per 16x16 tile ----
    const int h = wave;
    float* sS = (h < 6) ? (float*)(regQ + h * 16384) : (float*)(regS2 + (h - 6) * 16384);
    {
        const float* bh = biasb + h * (NTOK * NTOK);
        #pragma unroll
        for (int mt = 0; mt < 4; ++mt) {
            v16bf a = load_a_frag(s_qk + mt * 16 * 512 + h * 32, 512, 0, lane);
            #pragma unroll
            for (int nt = 0; nt < 4; ++nt) {
                v16bf b = load_b_frag(s_qk + nt * 16 * 512 + 256 + h * 32, 512, 0, lane);
                v8f acc = vzero8();
                acc = wmma_bf16(a, b, acc);
                #pragma unroll
                for (int r = 0; r < 8; ++r) {
                    int n = mt * 16 + r + hf * 8;
                    int m = nt * 16 + cl;
                    sS[n * NTOK + m] = acc[r] + bh[n * NTOK + m];
                }
            }
        }
    }
    __syncthreads();   // all waves done reading q/k -> safe to recycle as P

    // ---- softmax rows (2 rows/lane), P bf16 into recycled q/k region ----
    __bf16* sP = (__bf16*)(regK + h * 8192);             // [64][64] bf16
    {
        #pragma unroll
        for (int rr = 0; rr < 2; ++rr) {
            int n = lane * 2 + rr;
            const float* row = sS + n * NTOK;
            float mx = -3.4e38f;
            for (int m = 0; m < NTOK; ++m) mx = fmaxf(mx, row[m]);
            float sum = 0.0f;
            for (int m = 0; m < NTOK; ++m) sum += __expf(row[m] - mx);
            float inv = 1.0f / sum;
            for (int m = 0; m < NTOK; ++m)
                sP[n * NTOK + m] = (__bf16)(__expf(row[m] - mx) * inv);
        }
    }

    // ---- out_h[64,32] = P[64,64] @ v_h[64,32]  (vT rows are K-contiguous) ----
    {
        #pragma unroll
        for (int nt = 0; nt < 2; ++nt) {
            #pragma unroll
            for (int mt = 0; mt < 4; ++mt) {
                v8f acc = vzero8();
                #pragma unroll
                for (int kk = 0; kk < 2; ++kk) {
                    v16bf a = load_a_frag(sP + mt * 16 * NTOK, NTOK, kk * 32, lane);
                    v16bf b = load_b_frag(s_vT + (h * HD + nt * 16) * NTOK, NTOK, kk * 32, lane);
                    acc = wmma_bf16(a, b, acc);
                }
                #pragma unroll
                for (int r = 0; r < 8; ++r) {
                    int row = mt * 16 + r + hf * 8;
                    s_x[row * CDIM + h * HD + nt * 16 + cl] = (__bf16)acc[r]; // region X = attn out
                }
            }
        }
    }
    __syncthreads();

    // ---- GEMM3: out[64,256] = attn_out @ wp^T + proj_b -> global f32 ----
    {
        float* po = outp + (size_t)blk * (NTOK * CDIM);
        __builtin_prefetch(wp + wave * 2 * 16 * CDIM, 0, 3);
        #pragma unroll
        for (int i = 0; i < 8; ++i) {
            int mt = i >> 1;
            int nt = wave * 2 + (i & 1);
            v8f acc = vzero8();
            #pragma unroll
            for (int kk = 0; kk < 8; ++kk) {
                v16bf a = load_a_frag(s_x + mt * 16 * CDIM, CDIM, kk * 32, lane);
                v16bf b = load_b_frag(wp + nt * 16 * CDIM, CDIM, kk * 32, lane);
                acc = wmma_bf16(a, b, acc);
            }
            float pb = projb[nt * 16 + cl];
            #pragma unroll
            for (int r = 0; r < 8; ++r) {
                int row = mt * 16 + r + hf * 8;
                po[row * CDIM + nt * 16 + cl] = acc[r] + pb;
            }
        }
    }
}

// ---------------- host launch -----------------------------------------------
extern "C" void kernel_launch(void* const* d_in, const int* in_sizes, int n_in,
                              void* d_out, int out_size, void* d_ws, size_t ws_size,
                              hipStream_t stream) {
    const float* x    = (const float*)d_in[0];
    const float* qkvw = (const float*)d_in[1];
    const float* qkvb = (const float*)d_in[2];
    const float* dww  = (const float*)d_in[3];
    const float* dwb  = (const float*)d_in[4];
    const float* rpb  = (const float*)d_in[5];
    const float* temp = (const float*)d_in[6];
    const float* projw= (const float*)d_in[7];
    const float* projb= (const float*)d_in[8];

    // workspace layout
    __bf16* wq    = (__bf16*)d_ws;                                   // 768*256 bf16
    __bf16* wp    = (__bf16*)((char*)d_ws + TC * CDIM * 2);          // 256*256 bf16
    float*  biasb = (float*)((char*)d_ws + TC * CDIM * 2 + CDIM * CDIM * 2); // 8*64*64 f32

    int B = in_sizes[0] / (NTOK * CDIM);

    prep_kernel<<<(TC * CDIM + 255) / 256, 256, 0, stream>>>(qkvw, projw, rpb, wq, wp, biasb);

    size_t shmem = 32768 + 98304 + 98304 + 32768;   // 256 KB of the 320 KB WGP LDS
    winattn_kernel<<<B, 256, shmem, stream>>>(x, qkvb, dww, dwb, temp, projb,
                                              wq, wp, biasb, (float*)d_out);
}
